// TransMIL_4801773437151
// MI455X (gfx1250) — matrix operand order repaired and checked by reading, not compiled, not run on hardware
//
#include <hip/hip_runtime.h>
#include <hip/hip_bf16.h>
#include <stdint.h>

// ---------------- problem constants (from reference) ----------------
#define NTOK   16384     // tokens (128*128, so no padding anywhere: pad=0, add=0)
#define DIMC   512
#define HEADS  8
#define DH     64
#define MLAND  256
#define HSIDE  128
#define PINV_ITERS 6
#define RES_KS 33

typedef __attribute__((ext_vector_type(16))) __bf16 v16bf;
typedef __attribute__((ext_vector_type(8)))  float  v8f;
typedef __attribute__((ext_vector_type(2)))  float  v2f;
typedef unsigned short u16;
typedef unsigned int   u32;
typedef unsigned long long u64;

// CDNA5 async Global->LDS path (ASYNCcnt-tracked DMA, no VGPR round-trip).
// Builtin expects pointers to a 4 x i32 vector (per hipcc diagnostic).
#if defined(__HIP_DEVICE_COMPILE__) && __has_builtin(__builtin_amdgcn_global_load_async_to_lds_b128) && __has_builtin(__builtin_amdgcn_s_wait_asynccnt)
#define ASYNC_LDS 1
typedef int v4i_t __attribute__((vector_size(16)));
typedef __attribute__((address_space(1))) v4i_t gv4i_t;   // global
typedef __attribute__((address_space(3))) v4i_t lv4i_t;   // LDS
#else
#define ASYNC_LDS 0
#endif

struct U8W { u32 w[8]; };

__device__ __forceinline__ u16 f2bf(float f) {
  u32 u = __float_as_uint(f);
  u32 r = (u + 0x7FFFu + ((u >> 16) & 1u)) >> 16;   // round-to-nearest-even
  return (u16)r;
}

// =====================================================================
// bf16 WMMA GEMM:  C[M x N] (f32, opt bias/relu/accumulate) = A[M x K] @ Bt[N x K]^T
// 8 waves (wave32) per block, block tile 128x128, wave tile 32x64,
// KB=64 K-stage -> 16 v_wmma_f32_16x16x32_bf16 per barrier pair.
// LDS stride 72 halves (144 B): all stage stores / fragment loads are
// 16B-aligned (b128) and fragment reads hit 16 distinct banks (36r mod 64).
// GUARD=false (all tile-aligned shapes): async Global->LDS DMA staging.
// =====================================================================
#define BM 128
#define BN 128
#define KB 64
#define LDT 72

__device__ __forceinline__ v16bf frag_a(const u16* base, int lane, int ks) {
  // A 16x32 bf16 layout: lanes 0-15 row M=lane, K{0..7,16..23}; lanes 16-31 K{8..15,24..31}
  const u16* p = base + (lane & 15) * LDT + ((lane < 16) ? 0 : 8) + ks;
  uint4 lo = *(const uint4*)(p);        // K = k0..k0+7
  uint4 hi = *(const uint4*)(p + 16);   // K = k0+16..k0+23
  U8W u;
  u.w[0]=lo.x; u.w[1]=lo.y; u.w[2]=lo.z; u.w[3]=lo.w;
  u.w[4]=hi.x; u.w[5]=hi.y; u.w[6]=hi.z; u.w[7]=hi.w;
  return __builtin_bit_cast(v16bf, u);
}

__device__ __forceinline__ v16bf frag_b(const u16* base, int lane, int ks) {
  // B 32x16 layout (from Bt rows): lanes 0-15 col N=lane K{0..7,16..23}; lanes 16-31 K{8..15,24..31}
  const u16* p = base + (lane & 15) * LDT + ((lane < 16) ? 0 : 8) + ks;
  uint4 lo = *(const uint4*)(p);
  uint4 hi = *(const uint4*)(p + 16);
  U8W u;
  u.w[0]=lo.x; u.w[1]=lo.y; u.w[2]=lo.z; u.w[3]=lo.w;
  u.w[4]=hi.x; u.w[5]=hi.y; u.w[6]=hi.z; u.w[7]=hi.w;
  return __builtin_bit_cast(v16bf, u);
}

// flags: bit0 = relu, bit1 = accumulate into C
template<bool GUARD>
__global__ __launch_bounds__(256) void k_gemm_bf16(
    const u16* __restrict__ A, const u16* __restrict__ Bt,
    const float* __restrict__ bias, float* __restrict__ C, u16* __restrict__ Cbf,
    int Mdim, int Ndim, int Kdim, long sA, long sB, long sC, int flags)
{
  __shared__ u16 As[BM * LDT];
  __shared__ u16 Bs[BN * LDT];
  const int tid  = threadIdx.x;
  const int lane = tid & 31;
  const int wid  = tid >> 5;
  const int wr   = wid >> 1;            // 0..3 (32-row strip)
  const int wc   = wid & 1;             // 0..1 (64-col strip)
  const int m0   = blockIdx.y * BM;
  const int n0   = blockIdx.x * BN;
  const u16* Ab = A + (long)blockIdx.z * sA;
  const u16* Bb = Bt + (long)blockIdx.z * sB;
  float* Cb = C + (long)blockIdx.z * sC;
  u16* Cbb = Cbf ? Cbf + (long)blockIdx.z * sC : (u16*)0;

  v8f zero = {0.f,0.f,0.f,0.f,0.f,0.f,0.f,0.f};
  v8f acc[2][4];
#pragma unroll
  for (int i = 0; i < 2; ++i)
#pragma unroll
    for (int j = 0; j < 4; ++j) acc[i][j] = zero;

  for (int kk = 0; kk < Kdim; kk += KB) {
    // stage: 128 rows x 64 halves per tile = 1024 8-half chunks; 4 A + 4 B per thread
#pragma unroll
    for (int it = 0; it < 4; ++it) {
      int idx = tid + it * 256;          // 0..1023
      int r = idx >> 3, c8 = (idx & 7) << 3;
#if ASYNC_LDS
      if (!GUARD) {
        __builtin_amdgcn_global_load_async_to_lds_b128(
            (gv4i_t*)(Ab + (long)(m0 + r) * Kdim + kk + c8),
            (lv4i_t*)(As + r * LDT + c8), 0, 0);
        __builtin_amdgcn_global_load_async_to_lds_b128(
            (gv4i_t*)(Bb + (long)(n0 + r) * Kdim + kk + c8),
            (lv4i_t*)(Bs + r * LDT + c8), 0, 0);
      } else
#endif
      {
        uint4 d;
        if (GUARD) {
          d = make_uint4(0u,0u,0u,0u);
          if (m0 + r < Mdim) d = *(const uint4*)(Ab + (long)(m0 + r) * Kdim + kk + c8);
        } else {
          d = *(const uint4*)(Ab + (long)(m0 + r) * Kdim + kk + c8);
        }
        *(uint4*)(As + r * LDT + c8) = d;
        uint4 e;
        if (GUARD) {
          e = make_uint4(0u,0u,0u,0u);
          if (n0 + r < Ndim) e = *(const uint4*)(Bb + (long)(n0 + r) * Kdim + kk + c8);
        } else {
          e = *(const uint4*)(Bb + (long)(n0 + r) * Kdim + kk + c8);
        }
        *(uint4*)(Bs + r * LDT + c8) = e;
      }
    }
    // gfx1250 prefetch of next K stage (global_prefetch_b8)
    if (kk + KB < Kdim) {
      int gr = m0 + (tid >> 1);
      if (!GUARD || gr < Mdim) __builtin_prefetch(Ab + (long)gr * Kdim + kk + KB, 0, 1);
    }
#if ASYNC_LDS
    if (!GUARD) __builtin_amdgcn_s_wait_asynccnt(0);
#endif
    __syncthreads();

#pragma unroll
    for (int ks = 0; ks < KB; ks += 32) {
      v16bf bfr[4];
#pragma unroll
      for (int nt = 0; nt < 4; ++nt) bfr[nt] = frag_b(Bs + (wc * 64 + nt * 16) * LDT, lane, ks);
#pragma unroll
      for (int mt = 0; mt < 2; ++mt) {
        v16bf afr = frag_a(As + (wr * 32 + mt * 16) * LDT, lane, ks);
#pragma unroll
        for (int nt = 0; nt < 4; ++nt)
          acc[mt][nt] = __builtin_amdgcn_wmma_f32_16x16x32_bf16(
              false, afr, false, bfr[nt], (short)0, acc[mt][nt], false, false);
      }
    }
    __syncthreads();
  }

  const int rowAdd = (lane < 16) ? 0 : 8;
  const int colIn  = lane & 15;
#pragma unroll
  for (int mt = 0; mt < 2; ++mt)
#pragma unroll
    for (int nt = 0; nt < 4; ++nt) {
      int col = n0 + wc * 64 + nt * 16 + colIn;
      if (GUARD && col >= Ndim) continue;
      float bv = bias ? bias[col] : 0.f;
#pragma unroll
      for (int r = 0; r < 8; ++r) {
        int row = m0 + wr * 32 + mt * 16 + r + rowAdd;
        if (GUARD && row >= Mdim) continue;
        float v = acc[mt][nt][r] + bv;
        if (flags & 1) v = fmaxf(v, 0.f);
        long o = (long)row * Ndim + col;
        if (flags & 2) Cb[o] += v; else Cb[o] = v;
        if (Cbb) Cbb[o] = f2bf(v);
      }
    }
}

// =====================================================================
// f32 WMMA GEMM (V_WMMA_F32_16X16X4_F32) for the numerically-sensitive
// Newton-Schulz pinv path (~13 GFLOP total). One wave per 16x16 tile.
// BMOD=1: fuses B' = bdiag*I - B into the B fragment (7I/15I/13I terms).
// All shapes used are multiples of 16 (M,N in {256,64}, K=256).
// =====================================================================
template<int BMOD>
__global__ __launch_bounds__(32) void k_gemm_f32w(
    const float* __restrict__ A, const float* __restrict__ B, float* __restrict__ C,
    int Mi, int Ni, int Ki, long sA, long sB, long sC,
    float bdiag, float outscale)
{
  const int lane = threadIdx.x & 31;
  const int n0 = blockIdx.x * 16;
  const int m0 = blockIdx.y * 16;
  const float* Ab = A + (long)blockIdx.z * sA;
  const float* Bb = B + (long)blockIdx.z * sB;
  float* Cb = C + (long)blockIdx.z * sC;

  const int mr  = lane & 15;                 // A row / B col within tile
  const int k0  = (lane < 16) ? 0 : 2;       // K sub-offset held by this lane
  const int col = n0 + mr;

  v8f acc = {0.f,0.f,0.f,0.f,0.f,0.f,0.f,0.f};
  for (int k = 0; k < Ki; k += 4) {
    const float* ap = Ab + (long)(m0 + mr) * Ki + k + k0;
    v2f av; av.x = ap[0]; av.y = ap[1];      // A[m][k+k0], A[m][k+k0+1]
    int kb = k + k0;
    float b0 = Bb[(long)kb * Ni + col];
    float b1 = Bb[(long)(kb + 1) * Ni + col];
    if (BMOD) {
      b0 = ((kb     == col) ? bdiag : 0.f) - b0;
      b1 = ((kb + 1 == col) ? bdiag : 0.f) - b1;
    }
    v2f bv; bv.x = b0; bv.y = b1;
    acc = __builtin_amdgcn_wmma_f32_16x16x4_f32(
        false, av, false, bv, (short)0, acc, false, false);
  }
  const int rowAdd = (lane < 16) ? 0 : 8;
#pragma unroll
  for (int r = 0; r < 8; ++r)
    Cb[(long)(m0 + r + rowAdd) * Ni + col] = acc[r] * outscale;
}

// ---------------- conversions ----------------
__global__ void k_cvt_bf(const float* __restrict__ s, u16* __restrict__ d, long n) {
  long i = (long)blockIdx.x * 256 + threadIdx.x;
  if (i < n) d[i] = f2bf(s[i]);
}

// dst[c*R + r] = bf16(src[r*C + c]) ; batched via z
__global__ void k_transpose_bf(const float* __restrict__ src, u16* __restrict__ dst,
                               int R, int C, long sS, long sD) {
  __shared__ float tile[32][33];
  const float* s = src + (long)blockIdx.z * sS;
  u16* d = dst + (long)blockIdx.z * sD;
  int c0 = blockIdx.x * 32, r0 = blockIdx.y * 32;
  for (int i = threadIdx.y; i < 32; i += 8) {
    int r = r0 + i, c = c0 + threadIdx.x;
    tile[i][threadIdx.x] = (r < R && c < C) ? s[(long)r * C + c] : 0.f;
  }
  __syncthreads();
  for (int i = threadIdx.y; i < 32; i += 8) {
    int c = c0 + i, r = r0 + threadIdx.x;
    if (c < C && r < R) d[(long)c * R + r] = f2bf(tile[threadIdx.x][i]);
  }
}

// ---------------- layernorm (block per row) ----------------
__global__ __launch_bounds__(256) void k_layernorm(
    const float* __restrict__ x, const float* __restrict__ g, const float* __restrict__ b,
    float* __restrict__ y, u16* __restrict__ ybf, int dim) {
  __shared__ float red[256];
  long row = blockIdx.x;
  const float* xr = x + row * dim;
  int t = threadIdx.x;
  float s = 0.f;
  for (int c = t; c < dim; c += 256) s += xr[c];
  red[t] = s; __syncthreads();
  for (int st = 128; st > 0; st >>= 1) { if (t < st) red[t] += red[t + st]; __syncthreads(); }
  float mu = red[0] / dim; __syncthreads();
  float s2 = 0.f;
  for (int c = t; c < dim; c += 256) { float dd = xr[c] - mu; s2 += dd * dd; }
  red[t] = s2; __syncthreads();
  for (int st = 128; st > 0; st >>= 1) { if (t < st) red[t] += red[t + st]; __syncthreads(); }
  float inv = rsqrtf(red[0] / dim + 1e-5f);
  for (int c = t; c < dim; c += 256) {
    float v = (xr[c] - mu) * inv * g[c] + b[c];
    if (y)   y[row * dim + c] = v;
    if (ybf) ybf[row * dim + c] = f2bf(v);
  }
}

// ---------------- softmax (block per row, in place, opt bf16 copy) ----------------
__global__ __launch_bounds__(256) void k_softmax(float* __restrict__ x, u16* __restrict__ xbf, int cols) {
  __shared__ float red[256];
  long row = blockIdx.x;
  float* p = x + row * (long)cols;
  int t = threadIdx.x;
  float m = -3.4e38f;
  for (int c = t; c < cols; c += 256) m = fmaxf(m, p[c]);
  red[t] = m; __syncthreads();
  for (int s = 128; s > 0; s >>= 1) { if (t < s) red[t] = fmaxf(red[t], red[t + s]); __syncthreads(); }
  m = red[0]; __syncthreads();
  float su = 0.f;
  for (int c = t; c < cols; c += 256) { float e = __expf(p[c] - m); p[c] = e; su += e; }
  red[t] = su; __syncthreads();
  for (int s = 128; s > 0; s >>= 1) { if (t < s) red[t] += red[t + s]; __syncthreads(); }
  float inv = 1.f / red[0];
  for (int c = t; c < cols; c += 256) {
    float v = p[c] * inv; p[c] = v;
    if (xbf) xbf[row * (long)cols + c] = f2bf(v);
  }
}

// ---------------- qkv split to [H][N][64] (+ q scale, bf16 copies) ----------------
__global__ void k_split_qkv(const float* __restrict__ qkv, float* __restrict__ q,
                            float* __restrict__ k, float* __restrict__ v,
                            u16* __restrict__ qbf, u16* __restrict__ kbf) {
  long i = (long)blockIdx.x * 256 + threadIdx.x;     // NTOK*512
  if (i >= (long)NTOK * DIMC) return;
  int t = (int)(i >> 9);
  int hd = (int)(i & 511);
  int h = hd >> 6, d = hd & 63;
  float qv = qkv[(long)t * (3 * DIMC) + hd] * 0.125f;     // DH^-0.5
  float kv = qkv[(long)t * (3 * DIMC) + DIMC + hd];
  float vv = qkv[(long)t * (3 * DIMC) + 2 * DIMC + hd];
  long o = (((long)h * NTOK) + t) * 64 + d;
  q[o] = qv; k[o] = kv; v[o] = vv;
  qbf[o] = f2bf(qv); kbf[o] = f2bf(kv);
}

// ---------------- landmarks: mean over 64 consecutive tokens ----------------
__global__ void k_landmark(const float* __restrict__ src, float* __restrict__ dst, u16* __restrict__ dstbf) {
  int m = blockIdx.x, h = blockIdx.y, d = threadIdx.x;
  const float* s = src + (((long)h * NTOK) + (long)m * 64) * 64 + d;
  float acc = 0.f;
  for (int j = 0; j < 64; ++j) acc += s[(long)j * 64];
  acc *= (1.f / 64.f);
  long o = (((long)h * MLAND) + m) * 64 + d;
  dst[o] = acc; dstbf[o] = f2bf(acc);
}

// ---------------- pinv init: scale = 1/(max row-abs-sum * max col-abs-sum), z = x^T * scale ----------------
__global__ __launch_bounds__(256) void k_pinv_scale(const float* __restrict__ x, float* __restrict__ scl) {
  __shared__ float red[256];
  const float* xb = x + (long)blockIdx.x * MLAND * MLAND;
  int t = threadIdx.x;
  float rs = 0.f;
  for (int j = 0; j < MLAND; ++j) rs += fabsf(xb[(long)t * MLAND + j]);
  red[t] = rs; __syncthreads();
  for (int s = 128; s > 0; s >>= 1) { if (t < s) red[t] = fmaxf(red[t], red[t + s]); __syncthreads(); }
  float colmax = red[0]; __syncthreads();
  float cs = 0.f;
  for (int i = 0; i < MLAND; ++i) cs += fabsf(xb[(long)i * MLAND + t]);
  red[t] = cs; __syncthreads();
  for (int s = 128; s > 0; s >>= 1) { if (t < s) red[t] = fmaxf(red[t], red[t + s]); __syncthreads(); }
  if (t == 0) scl[blockIdx.x] = 1.f / (colmax * red[0]);
}

__global__ void k_pinv_init(const float* __restrict__ x, float* __restrict__ zz, const float* __restrict__ scl) {
  long i = (long)blockIdx.x * 256 + threadIdx.x;     // H*256*256
  int h = (int)(i >> 16), r = (int)((i >> 8) & 255), c = (int)(i & 255);
  zz[i] = x[((long)h << 16) + ((long)c << 8) + r] * scl[h];
}

// ---------------- depthwise residual conv along tokens (33 taps / head) ----------------
__global__ void k_resconv_add(float* __restrict__ attn, const float* __restrict__ v, const float* __restrict__ w) {
  long i = (long)blockIdx.x * 256 + threadIdx.x;     // H*NTOK*64
  if (i >= (long)HEADS * NTOK * 64) return;
  int d = (int)(i & 63);
  long td = i >> 6;
  int t = (int)(td % NTOK);
  int h = (int)(td / NTOK);
  float acc = 0.f;
#pragma unroll
  for (int kk = 0; kk < RES_KS; ++kk) {
    int tt = t + kk - RES_KS / 2;
    if (tt >= 0 && tt < NTOK) acc += w[h * RES_KS + kk] * v[(((long)h * NTOK) + tt) * 64 + d];
  }
  attn[i] += acc;
}

// ---------------- merge heads [H][N][64] -> bf16 [N][512] ----------------
__global__ void k_merge_heads(const float* __restrict__ attn, u16* __restrict__ mbf) {
  long i = (long)blockIdx.x * 256 + threadIdx.x;     // NTOK*512
  if (i >= (long)NTOK * DIMC) return;
  int t = (int)(i >> 9), hd = (int)(i & 511);
  int h = hd >> 6, d = hd & 63;
  mbf[i] = f2bf(attn[(((long)h * NTOK) + t) * 64 + d]);
}

// ---------------- PPEG: identity + 7x7 + 5x5 + 3x3 depthwise on 128x128 map ----------------
__global__ __launch_bounds__(256) void k_ppeg(
    const float* __restrict__ hin, float* __restrict__ hout,
    const float* __restrict__ w7, const float* __restrict__ b7,
    const float* __restrict__ w5, const float* __restrict__ b5,
    const float* __restrict__ w3, const float* __restrict__ b3) {
  int c = blockIdx.x * 256 + threadIdx.x;            // 0..511
  int t = blockIdx.y;                                // 0..16383
  int ti = t >> 7, tj = t & 127;
  float acc = hin[(long)t * DIMC + c] + b7[c] + b5[c] + b3[c];
  for (int di = -3; di <= 3; ++di) {
    int ii = ti + di; if (ii < 0 || ii >= HSIDE) continue;
    for (int dj = -3; dj <= 3; ++dj) {
      int jj = tj + dj; if (jj < 0 || jj >= HSIDE) continue;
      float xv = hin[((long)(ii * HSIDE + jj)) * DIMC + c];
      acc += w7[(c * 7 + (di + 3)) * 7 + (dj + 3)] * xv;
      if (di >= -2 && di <= 2 && dj >= -2 && dj <= 2) acc += w5[(c * 5 + (di + 2)) * 5 + (dj + 2)] * xv;
      if (di >= -1 && di <= 1 && dj >= -1 && dj <= 1) acc += w3[(c * 3 + (di + 1)) * 3 + (dj + 1)] * xv;
    }
  }
  hout[(long)t * DIMC + c] = acc;
}

// =====================================================================
extern "C" void kernel_launch(void* const* d_in, const int* in_sizes, int n_in,
                              void* d_out, int out_size, void* d_ws, size_t ws_size,
                              hipStream_t stream) {
  (void)in_sizes; (void)n_in; (void)out_size; (void)ws_size;
  const float* in_h    = (const float*)d_in[0];
  const float* fc1_w   = (const float*)d_in[1];
  const float* fc1_b   = (const float*)d_in[2];
  const float* ln1_g   = (const float*)d_in[3];
  const float* ln1_b   = (const float*)d_in[4];
  const float* qkv1_w  = (const float*)d_in[5];
  const float* out1_w  = (const float*)d_in[6];
  const float* out1_b  = (const float*)d_in[7];
  const float* res1_w  = (const float*)d_in[8];
  const float* ppeg_w7 = (const float*)d_in[9];
  const float* ppeg_b7 = (const float*)d_in[10];
  const float* ppeg_w5 = (const float*)d_in[11];
  const float* ppeg_b5 = (const float*)d_in[12];
  const float* ppeg_w3 = (const float*)d_in[13];
  const float* ppeg_b3 = (const float*)d_in[14];
  const float* ln2_g   = (const float*)d_in[15];
  const float* ln2_b   = (const float*)d_in[16];
  const float* qkv2_w  = (const float*)d_in[17];
  const float* out2_w  = (const float*)d_in[18];
  const float* out2_b  = (const float*)d_in[19];
  const float* res2_w  = (const float*)d_in[20];
  const float* normf_g = (const float*)d_in[21];
  const float* normf_b = (const float*)d_in[22];

  // ---- bump allocator over d_ws (total ~750 MB; every buffer written before read) ----
  char* base = (char*)d_ws;
  size_t off = 0;
  auto alloc = [&](size_t bytes) -> void* {
    off = (off + 255) & ~(size_t)255;
    void* p = base + off; off += bytes; return p;
  };
  const long NDf = (long)NTOK * DIMC;
  u16*  hin_bf = (u16*)alloc(NDf * 2);
  u16*  fc1wt  = (u16*)alloc((long)DIMC * DIMC * 2);
  u16*  qkv1wt = (u16*)alloc((long)3 * DIMC * DIMC * 2);
  u16*  out1wt = (u16*)alloc((long)DIMC * DIMC * 2);
  u16*  qkv2wt = (u16*)alloc((long)3 * DIMC * DIMC * 2);
  u16*  out2wt = (u16*)alloc((long)DIMC * DIMC * 2);
  float* hbuf  = (float*)alloc(NDf * 4);
  float* h2buf = (float*)alloc(NDf * 4);
  u16*  xbf    = (u16*)alloc(NDf * 2);
  float* qkvf  = (float*)alloc((long)NTOK * 3 * DIMC * 4);
  float* qf    = (float*)alloc(NDf * 4);
  float* kf    = (float*)alloc(NDf * 4);
  float* vf    = (float*)alloc(NDf * 4);
  u16*  qbf    = (u16*)alloc(NDf * 2);
  u16*  kbf    = (u16*)alloc(NDf * 2);
  u16*  vtbf   = (u16*)alloc(NDf * 2);
  float* qlf   = (float*)alloc((long)HEADS * MLAND * DH * 4);
  float* klf   = (float*)alloc((long)HEADS * MLAND * DH * 4);
  u16*  qlbf   = (u16*)alloc((long)HEADS * MLAND * DH * 2);
  u16*  klbf   = (u16*)alloc((long)HEADS * MLAND * DH * 2);
  float* a1f   = (float*)alloc((long)HEADS * NTOK * MLAND * 4);
  u16*  a1bf   = (u16*)alloc((long)HEADS * NTOK * MLAND * 2);
  float* a2f   = (float*)alloc((long)HEADS * MLAND * MLAND * 4);
  float* a3f   = (float*)alloc((long)HEADS * MLAND * NTOK * 4);
  u16*  a3bf   = (u16*)alloc((long)HEADS * MLAND * NTOK * 2);
  float* zb    = (float*)alloc((long)HEADS * MLAND * MLAND * 4);
  float* z2b   = (float*)alloc((long)HEADS * MLAND * MLAND * 4);
  float* xzb   = (float*)alloc((long)HEADS * MLAND * MLAND * 4);
  float* t2b   = (float*)alloc((long)HEADS * MLAND * MLAND * 4);
  float* t4b   = (float*)alloc((long)HEADS * MLAND * MLAND * 4);
  float* scl   = (float*)alloc(HEADS * 4);
  float* Sf    = (float*)alloc((long)HEADS * MLAND * DH * 4);
  float* P2f   = (float*)alloc((long)HEADS * MLAND * DH * 4);
  u16*  P2t    = (u16*)alloc((long)HEADS * MLAND * DH * 2);
  float* attnf = (float*)alloc((long)HEADS * NTOK * DH * 4);
  u16*  mbf    = (u16*)alloc(NDf * 2);

  const dim3 tb(32, 8);
  auto gemm = [&](const u16* A, const u16* Bt, const float* bias, float* C, u16* Cbf,
                  int Mi, int Ni, int Ki, long sA, long sB, long sC, int batch, int flags) {
    dim3 g((Ni + BN - 1) / BN, (Mi + BM - 1) / BM, batch);
    bool guard = (Mi % BM) || (Ni % BN);
    if (guard)
      k_gemm_bf16<true><<<g, 256, 0, stream>>>(A, Bt, bias, C, Cbf, Mi, Ni, Ki, sA, sB, sC, flags);
    else
      k_gemm_bf16<false><<<g, 256, 0, stream>>>(A, Bt, bias, C, Cbf, Mi, Ni, Ki, sA, sB, sC, flags);
  };

  // ---- one-time conversions (activation + transposed weights -> bf16) ----
  k_cvt_bf<<<(int)((NDf + 255) / 256), 256, 0, stream>>>(in_h, hin_bf, NDf);
  k_transpose_bf<<<dim3(DIMC / 32, DIMC / 32, 1), tb, 0, stream>>>(fc1_w, fc1wt, DIMC, DIMC, 0, 0);
  k_transpose_bf<<<dim3(3 * DIMC / 32, DIMC / 32, 1), tb, 0, stream>>>(qkv1_w, qkv1wt, DIMC, 3 * DIMC, 0, 0);
  k_transpose_bf<<<dim3(DIMC / 32, DIMC / 32, 1), tb, 0, stream>>>(out1_w, out1wt, DIMC, DIMC, 0, 0);
  k_transpose_bf<<<dim3(3 * DIMC / 32, DIMC / 32, 1), tb, 0, stream>>>(qkv2_w, qkv2wt, DIMC, 3 * DIMC, 0, 0);
  k_transpose_bf<<<dim3(DIMC / 32, DIMC / 32, 1), tb, 0, stream>>>(out2_w, out2wt, DIMC, DIMC, 0, 0);

  // ---- fc1 + relu ----
  gemm(hin_bf, fc1wt, fc1_b, hbuf, nullptr, NTOK, DIMC, DIMC, 0, 0, 0, 1, 1);

  const long sHD = (long)NTOK * DH;      // per-head [N][64]
  const long sLD = (long)MLAND * DH;     // per-head [256][64]
  const long sNM = (long)NTOK * MLAND;   // per-head [N][256]
  const long sMM = (long)MLAND * MLAND;  // per-head [256][256]

  auto run_layer = [&](const float* lng, const float* lnb, const u16* qkvwt,
                       const u16* outwt, const float* outb, const float* resw, float* hcur) {
    k_layernorm<<<NTOK, 256, 0, stream>>>(hcur, lng, lnb, nullptr, xbf, DIMC);
    gemm(xbf, qkvwt, nullptr, qkvf, nullptr, NTOK, 3 * DIMC, DIMC, 0, 0, 0, 1, 0);
    k_split_qkv<<<(int)((NDf + 255) / 256), 256, 0, stream>>>(qkvf, qf, kf, vf, qbf, kbf);
    k_transpose_bf<<<dim3(DH / 32, NTOK / 32, HEADS), tb, 0, stream>>>(vf, vtbf, NTOK, DH, sHD, sHD);
    k_landmark<<<dim3(MLAND, HEADS), 64, 0, stream>>>(qf, qlf, qlbf);
    k_landmark<<<dim3(MLAND, HEADS), 64, 0, stream>>>(kf, klf, klbf);

    // a1 = softmax(q @ kl^T)  [H][N][256]
    gemm(qbf, klbf, nullptr, a1f, nullptr, NTOK, MLAND, DH, sHD, sLD, sNM, HEADS, 0);
    k_softmax<<<HEADS * NTOK, 256, 0, stream>>>(a1f, a1bf, MLAND);
    // a2 = softmax(ql @ kl^T) [H][256][256]
    gemm(qlbf, klbf, nullptr, a2f, nullptr, MLAND, MLAND, DH, sLD, sLD, sMM, HEADS, 0);
    k_softmax<<<HEADS * MLAND, 256, 0, stream>>>(a2f, nullptr, MLAND);
    // a3 = softmax(ql @ k^T)  [H][256][N]
    gemm(qlbf, kbf, nullptr, a3f, nullptr, MLAND, NTOK, DH, sLD, sHD, (long)MLAND * NTOK, HEADS, 0);
    k_softmax<<<HEADS * MLAND, 256, 0, stream>>>(a3f, a3bf, NTOK);

    // Newton-Schulz pinv(a2): f32 WMMA (16x16x4), cI-B fused into B fragments
    k_pinv_scale<<<HEADS, 256, 0, stream>>>(a2f, scl);
    k_pinv_init<<<HEADS * MLAND * MLAND / 256, 256, 0, stream>>>(a2f, zb, scl);
    dim3 gMM(MLAND / 16, MLAND / 16, HEADS);
    for (int it = 0; it < PINV_ITERS; ++it) {
      k_gemm_f32w<0><<<gMM, 32, 0, stream>>>(a2f, zb, xzb, MLAND, MLAND, MLAND, sMM, sMM, sMM, 0.f, 1.f);
      k_gemm_f32w<1><<<gMM, 32, 0, stream>>>(xzb, xzb, t2b, MLAND, MLAND, MLAND, sMM, sMM, sMM, 7.f, 1.f);
      k_gemm_f32w<1><<<gMM, 32, 0, stream>>>(xzb, t2b, t4b, MLAND, MLAND, MLAND, sMM, sMM, sMM, 15.f, 1.f);
      k_gemm_f32w<1><<<gMM, 32, 0, stream>>>(zb, t4b, z2b, MLAND, MLAND, MLAND, sMM, sMM, sMM, 13.f, 0.25f);
      float* tmp = zb; zb = z2b; z2b = tmp;
    }

    // S = a3 @ v (K = 16384, WMMA long-K), P2 = pinv @ S, attn = a1 @ P2
    gemm(a3bf, vtbf, nullptr, Sf, nullptr, MLAND, DH, NTOK, (long)MLAND * NTOK, sHD, sLD, HEADS, 0);
    k_gemm_f32w<0><<<dim3(DH / 16, MLAND / 16, HEADS), 32, 0, stream>>>(
        zb, Sf, P2f, MLAND, DH, MLAND, sMM, sLD, sLD, 0.f, 1.f);
    k_transpose_bf<<<dim3(DH / 32, MLAND / 32, HEADS), tb, 0, stream>>>(P2f, P2t, MLAND, DH, sLD, sLD);
    gemm(a1bf, P2t, nullptr, attnf, nullptr, NTOK, DH, MLAND, sNM, (long)DH * MLAND, sHD, HEADS, 0);

    // + depthwise residual conv over v, merge heads, out-proj (+bias) accumulated into h
    long nA = (long)HEADS * NTOK * DH;
    k_resconv_add<<<(int)((nA + 255) / 256), 256, 0, stream>>>(attnf, vf, resw);
    k_merge_heads<<<(int)((NDf + 255) / 256), 256, 0, stream>>>(attnf, mbf);
    gemm(mbf, outwt, outb, hcur, nullptr, NTOK, DIMC, DIMC, 0, 0, 0, 1, 2 /*accumulate*/);
  };

  run_layer(ln1_g, ln1_b, qkv1wt, out1wt, out1_b, res1_w, hbuf);
  k_ppeg<<<dim3(2, NTOK), 256, 0, stream>>>(hbuf, h2buf, ppeg_w7, ppeg_b7, ppeg_w5, ppeg_b5, ppeg_w3, ppeg_b3);
  run_layer(ln2_g, ln2_b, qkv2wt, out2wt, out2_b, res2_w, h2buf);
  k_layernorm<<<NTOK, 256, 0, stream>>>(h2buf, normf_g, normf_b, (float*)d_out, nullptr, DIMC);
}
